// PLE_26010321944857
// MI455X (gfx1250) — matrix operand order repaired and checked by reading, not compiled, not run on hardware
//
#include <hip/hip_runtime.h>
#include <stdint.h>

// ---------------------------------------------------------------------------
// PLE forward for MI455X (gfx1250): all large matmuls via
// V_WMMA_F32_16X16X32_BF16 (fp32 in -> bf16 tiles in LDS -> fp32 accum).
// 128x128x32 block tile, 8 wave32s (each wave 32x64 = 2x4 WMMA accums),
// double-buffered LDS: one barrier per K-step, 8 WMMAs per wave per step.
// ---------------------------------------------------------------------------

typedef __bf16 bf16_t;
typedef __attribute__((ext_vector_type(16))) __bf16 v16bf;
typedef __attribute__((ext_vector_type(8)))  __bf16 v8bf;
typedef __attribute__((ext_vector_type(8)))  float  v8f;
typedef __attribute__((ext_vector_type(4)))  float  v4f;

#define BM 128
#define BN 128
#define BK 32
#define LDSP (BK + 8)   // row stride 40 bf16 = 80B (16B-aligned rows)

// C[z] = act( A[z/aDiv] (MxK) @ W[z] (KxN) + bias[z] ), all fp32 row-major.
template<int ACT>
__global__ __launch_bounds__(256)
void gemm_bias_kernel(const float* __restrict__ A, long long aZ, int aDiv,
                      const float* __restrict__ W, long long wZ,
                      const float* __restrict__ bias, long long bZ,
                      float* __restrict__ C, long long cZ,
                      int M, int N, int K)
{
  __shared__ bf16_t sA[2][BM][LDSP];   // [buf][m][k]
  __shared__ bf16_t sB[2][BN][LDSP];   // [buf][n][k] (W tile stored transposed)

  const int z = blockIdx.z;
  const float* Ap = A + (long long)(z / aDiv) * aZ;
  const float* Wp = W + (long long)z * wZ;
  const float* bp = bias ? bias + (long long)z * bZ : nullptr;
  float*       Cp = C + (long long)z * cZ;

  const int m0 = blockIdx.y * BM;
  const int n0 = blockIdx.x * BN;

  const int tid  = threadIdx.x;
  const int lane = tid & 31;
  const int wv   = tid >> 5;
  const int wm   = (wv & 3) * 32;   // wave row offset (4 waves over M)
  const int wn   = (wv >> 2) * 64;  // wave col offset (2 waves over N)
  const int l16  = lane & 15;
  const int half = lane >> 4;       // 0: lanes 0-15, 1: lanes 16-31

  v8f acc[2][4] = {};

  // stage one BMxBK A tile and BKxBN W tile (transposed) into LDS buffer p
  auto stage = [&](int p, int k0) {
    // A: 128x32 fp32 = 1024 float4, 4 per thread
    #pragma unroll
    for (int i = 0; i < 4; ++i) {
      int lin = tid + i * 256;
      int row = lin >> 3;                  // 8 float4 per 32-float row
      int c4  = (lin & 7) << 2;
      int gm  = m0 + row;
      v4f a;
      if (gm < M) a = *(const v4f*)(Ap + (long long)gm * K + k0 + c4);
      else        a = (v4f){0.f, 0.f, 0.f, 0.f};
      bf16_t* dst = &sA[p][row][c4];
      dst[0] = (bf16_t)a.x; dst[1] = (bf16_t)a.y;
      dst[2] = (bf16_t)a.z; dst[3] = (bf16_t)a.w;
    }
    // W: 32x128 fp32 = 1024 float4, 4 per thread; store transposed [n][k]
    #pragma unroll
    for (int i = 0; i < 4; ++i) {
      int lin = tid + i * 256;
      int kr  = lin >> 5;                  // 32 float4 per 128-float row
      int c4  = (lin & 31) << 2;
      int gn  = n0 + c4;
      const float* wrow = Wp + (long long)(k0 + kr) * N;
      v4f w;
      if (gn + 3 < N) {
        w = *(const v4f*)(wrow + gn);
      } else {
        w.x = (gn + 0 < N) ? wrow[gn + 0] : 0.f;
        w.y = (gn + 1 < N) ? wrow[gn + 1] : 0.f;
        w.z = (gn + 2 < N) ? wrow[gn + 2] : 0.f;
        w.w = (gn + 3 < N) ? wrow[gn + 3] : 0.f;
      }
      sB[p][c4 + 0][kr] = (bf16_t)w.x; sB[p][c4 + 1][kr] = (bf16_t)w.y;
      sB[p][c4 + 2][kr] = (bf16_t)w.z; sB[p][c4 + 3][kr] = (bf16_t)w.w;
    }
  };

  // compute 8 WMMAs from LDS buffer p (fragment layouts per CDNA5 ISA):
  //  A 16x32: lane l -> row M=(l&15); K = base..base+7, base+16..base+23,
  //           base = (l>=16)?8:0   (two 16B LDS reads)
  //  B 32x16: lane l -> col N=(l&15); same K order as A
  auto compute = [&](int p) {
    v16bf aF[2], bF[4];
    #pragma unroll
    for (int r = 0; r < 2; ++r) {
      const bf16_t* src = &sA[p][wm + r * 16 + l16][half * 8];
      union { v16bf v; v8bf h[2]; } u;
      u.h[0] = *(const v8bf*)(src);
      u.h[1] = *(const v8bf*)(src + 16);
      aF[r] = u.v;
    }
    #pragma unroll
    for (int c = 0; c < 4; ++c) {
      const bf16_t* src = &sB[p][wn + c * 16 + l16][half * 8];
      union { v16bf v; v8bf h[2]; } u;
      u.h[0] = *(const v8bf*)(src);
      u.h[1] = *(const v8bf*)(src + 16);
      bF[c] = u.v;
    }
    #pragma unroll
    for (int r = 0; r < 2; ++r)
      #pragma unroll
      for (int c = 0; c < 4; ++c)
        acc[r][c] = __builtin_amdgcn_wmma_f32_16x16x32_bf16(
            false, aF[r], false, bF[c], (short)0, acc[r][c], false, false);
  };

  const int nk = K / BK;   // all K dims are multiples of 32
  stage(0, 0);
  __syncthreads();
  for (int i = 0; i < nk; ++i) {
    int p = i & 1;
    if (i + 1 < nk) {
      stage(p ^ 1, (i + 1) * BK);
      __builtin_prefetch(Ap + (long long)(m0 + (tid >> 1)) * K + (i + 2) * BK, 0, 1);
      __builtin_prefetch(Wp + (long long)((i + 2) * BK + (tid >> 3)) * N + n0 + (tid & 7) * 16, 0, 1);
    }
    compute(p);
    __syncthreads();
  }

  // epilogue: D layout lane l -> N=(l&15), M=(l>=16?8:0)+v
  #pragma unroll
  for (int c = 0; c < 4; ++c) {
    int gn = n0 + wn + c * 16 + l16;
    if (gn >= N) continue;
    float bv = bp ? bp[gn] : 0.f;
    #pragma unroll
    for (int r = 0; r < 2; ++r) {
      int gmBase = m0 + wm + r * 16 + half * 8;
      #pragma unroll
      for (int v = 0; v < 8; ++v) {
        int gm = gmBase + v;
        if (gm < M) {
          float x = acc[r][c][v] + bv;
          if (ACT) x = x > 0.f ? x : 0.f;
          Cp[(long long)gm * N + gn] = x;
        }
      }
    }
  }
}

// In-place row softmax, E <= 16.
__global__ void softmax_rows(float* __restrict__ x, long long rows, int E) {
  long long r = (long long)blockIdx.x * blockDim.x + threadIdx.x;
  if (r >= rows) return;
  float* p = x + r * E;
  float buf[16];
  float mx = -3.0e38f;
  for (int i = 0; i < E; ++i) { buf[i] = p[i]; mx = fmaxf(mx, buf[i]); }
  float s = 0.f;
  for (int i = 0; i < E; ++i) { buf[i] = __expf(buf[i] - mx); s += buf[i]; }
  float inv = 1.f / s;
  for (int i = 0; i < E; ++i) p[i] = buf[i] * inv;
}

// out[t,b,o] = sum_s g[t,b,s]*sh[s,b,o] + sum_k g[t,b,4+k]*tk[t*4+k,b,o]
__global__ void mix_tasks_kernel(const float* __restrict__ g,   // (3,B,8)
                                 const float* __restrict__ sh,  // (4,B,O)
                                 const float* __restrict__ tk,  // (12,B,O)
                                 float* __restrict__ out,       // (3,B,O)
                                 int Bn, int Od) {
  long long idx = (long long)blockIdx.x * blockDim.x + threadIdx.x;
  long long total = (long long)3 * Bn * Od;
  if (idx >= total) return;
  int o = (int)(idx % Od);
  long long bt = idx / Od;
  int b = (int)(bt % Bn);
  int t = (int)(bt / Bn);
  const float* gr = g + ((long long)t * Bn + b) * 8;
  float acc = 0.f;
  #pragma unroll
  for (int s = 0; s < 4; ++s)
    acc += gr[s] * sh[((long long)s * Bn + b) * Od + o];
  #pragma unroll
  for (int k = 0; k < 4; ++k)
    acc += gr[4 + k] * tk[((long long)(t * 4 + k) * Bn + b) * Od + o];
  out[((long long)t * Bn + b) * Od + o] = acc;
}

// out[b,o] = sum_{e<12} gl[b,e]*tk[e,b,o] + sum_{e<4} gl[b,12+e]*sh[e,b,o]
__global__ void mix_last_kernel(const float* __restrict__ gl,  // (B,16)
                                const float* __restrict__ tk,  // (12,B,O)
                                const float* __restrict__ sh,  // (4,B,O)
                                float* __restrict__ out,       // (B,O)
                                int Bn, int Od) {
  long long idx = (long long)blockIdx.x * blockDim.x + threadIdx.x;
  long long total = (long long)Bn * Od;
  if (idx >= total) return;
  int o = (int)(idx % Od);
  int b = (int)(idx / Od);
  const float* gr = gl + (long long)b * 16;
  float acc = 0.f;
  #pragma unroll
  for (int e = 0; e < 12; ++e)
    acc += gr[e] * tk[((long long)e * Bn + b) * Od + o];
  #pragma unroll
  for (int e = 0; e < 4; ++e)
    acc += gr[12 + e] * sh[((long long)e * Bn + b) * Od + o];
  out[(long long)b * Od + o] = acc;
}

extern "C" void kernel_launch(void* const* d_in, const int* in_sizes, int n_in,
                              void* d_out, int out_size, void* d_ws, size_t ws_size,
                              hipStream_t stream) {
  (void)in_sizes; (void)n_in; (void)out_size; (void)ws_size;
  const int B = 4096, D = 512, H = 1024, O = 512;

  const float* in[35];
  for (int i = 0; i < 35; ++i) in[i] = (const float*)d_in[i];
  const float* user = in[0];
  float* outp = (float*)d_out;

  // ---- workspace layout (floats); total ~92.4M floats (~370 MB) ----
  float* ws = (float*)d_ws;
  size_t off = 0;
  float* hid   = ws + off; off += (size_t)12 * B * H;  // expert hiddens (reused)
  float* sh1   = ws + off; off += (size_t)4  * B * O;  // sh1, later sh2
  float* tk1   = ws + off; off += (size_t)12 * B * O;  // tk1, later tk2
  float* g1    = ws + off; off += (size_t)3  * B * 8;  // g1, later g2
  float* g1l   = ws + off; off += (size_t)B * 16;
  float* tfeat = ws + off; off += (size_t)3  * B * O;  // task_feats, later tower_in
  float* flast = ws + off; off += (size_t)B * O;

  auto gemm = [&](const float* A, long long aZ, int aDiv,
                  const float* W, long long wZ,
                  const float* bias, long long bZ,
                  float* C, long long cZ,
                  int M, int N, int Kd, int Z, bool relu) {
    dim3 grid((N + BN - 1) / BN, (M + BM - 1) / BM, Z);
    if (relu)
      gemm_bias_kernel<1><<<grid, 256, 0, stream>>>(A, aZ, aDiv, W, wZ, bias, bZ, C, cZ, M, N, Kd);
    else
      gemm_bias_kernel<0><<<grid, 256, 0, stream>>>(A, aZ, aDiv, W, wZ, bias, bZ, C, cZ, M, N, Kd);
  };

  // ---- Level-1 experts ----
  gemm(user, 0, 1, in[1], (long long)D * H, in[2], H, hid, (long long)B * H, B, H, D, 4,  true);
  gemm(hid, (long long)B * H, 1, in[3], (long long)H * O, in[4], O, sh1, (long long)B * O, B, O, H, 4,  false);
  gemm(user, 0, 1, in[5], (long long)D * H, in[6], H, hid, (long long)B * H, B, H, D, 12, true);
  gemm(hid, (long long)B * H, 1, in[7], (long long)H * O, in[8], O, tk1, (long long)B * O, B, O, H, 12, false);

  // ---- Level-1 gates ----
  gemm(user, 0, 1, in[9],  (long long)D * 8, in[10], 8, g1,  (long long)B * 8, B, 8,  D, 3, false);
  softmax_rows<<<(3 * B + 255) / 256, 256, 0, stream>>>(g1, (long long)3 * B, 8);
  gemm(user, 0, 1, in[11], 0,               in[12], 0, g1l, 0,                B, 16, D, 1, false);
  softmax_rows<<<(B + 255) / 256, 256, 0, stream>>>(g1l, B, 16);

  // ---- Level-1 mixing ----
  long long tot1 = (long long)3 * B * O;
  mix_tasks_kernel<<<(unsigned)((tot1 + 255) / 256), 256, 0, stream>>>(g1, sh1, tk1, tfeat, B, O);
  long long tot2 = (long long)B * O;
  mix_last_kernel<<<(unsigned)((tot2 + 255) / 256), 256, 0, stream>>>(g1l, tk1, sh1, flast, B, O);

  // ---- Level-2 experts (overwrite sh1 -> sh2, tk1 -> tk2) ----
  gemm(flast, 0, 1, in[13], (long long)O * H, in[14], H, hid, (long long)B * H, B, H, O, 4,  true);
  gemm(hid, (long long)B * H, 1, in[15], (long long)H * O, in[16], O, sh1, (long long)B * O, B, O, H, 4,  false);
  gemm(tfeat, (long long)B * O, 4, in[17], (long long)O * H, in[18], H, hid, (long long)B * H, B, H, O, 12, true);
  gemm(hid, (long long)B * H, 1, in[19], (long long)H * O, in[20], O, tk1, (long long)B * O, B, O, H, 12, false);

  // ---- Level-2 gates (overwrite g1 -> g2) ----
  gemm(tfeat, (long long)B * O, 1, in[21], (long long)O * 8, in[22], 8, g1, (long long)B * 8, B, 8, O, 3, false);
  softmax_rows<<<(3 * B + 255) / 256, 256, 0, stream>>>(g1, (long long)3 * B, 8);

  // ---- Level-2 mixing -> tower_in (overwrite tfeat) ----
  mix_tasks_kernel<<<(unsigned)((tot1 + 255) / 256), 256, 0, stream>>>(g1, sh1, tk1, tfeat, B, O);

  // ---- Towers (hidden region is free now; carve from it) ----
  float* th1 = hid;
  float* th2 = th1 + (size_t)B * 512;
  float* bh1 = th2 + (size_t)B * 256;
  float* bh2 = bh1 + (size_t)2 * B * 512;

  // churn tower: 512 -> 512 -> 256 -> 1, output at d_out[0..B)
  gemm(tfeat, 0, 1, in[23], 0, in[24], 0, th1, 0, B, 512, O,   1, true);
  gemm(th1,   0, 1, in[25], 0, in[26], 0, th2, 0, B, 256, 512, 1, true);
  gemm(th2,   0, 1, in[27], 0, in[28], 0, outp, 0, B, 1, 256,  1, false);

  // bt towers for tasks 1,2: 512 -> 512 -> 256 -> 100, outputs at d_out[B + z*B*100 ...]
  gemm(tfeat + (size_t)B * O, (long long)B * O, 1, in[29], (long long)O * 512, in[30], 512,
       bh1, (long long)B * 512, B, 512, O, 2, true);
  gemm(bh1, (long long)B * 512, 1, in[31], (long long)512 * 256, in[32], 256,
       bh2, (long long)B * 256, B, 256, 512, 2, true);
  gemm(bh2, (long long)B * 256, 1, in[33], (long long)256 * 100, in[34], 100,
       outp + B, (long long)B * 100, B, 100, 256, 2, false);
}
